// GalerkinAttention3d_44169443672700
// MI455X (gfx1250) — matrix operand order repaired and checked by reading, not compiled, not run on hardware
//
#include <hip/hip_runtime.h>

// Galerkin (linear) attention, MI455X / gfx1250, wave32 + WMMA bf16.
// Pipeline: cvt/swizzle -> fused proj+LN (Q, K^T, V^T bf16) -> split-K K^TV
//           (f32 atomics, double-buffered LDS) -> P_h = M_h @ Wo_h (swizzled)
//           -> out = sum_h Q_h P_h (double-buffered Q tiles).
// Workspace use ~= 422 MB.

typedef __attribute__((ext_vector_type(16))) __bf16 v16bf;
typedef __attribute__((ext_vector_type(8)))  float  v8f;

#define DEV __device__ __forceinline__

constexpr int Bz = 2, Hh = 8, Ll = 32768, Cin = 128, Dd = 128, HDim = 1024;

DEV unsigned short f2bf(float f) {
  unsigned u = __float_as_uint(f);
  return (unsigned short)((u + 0x7FFFu + ((u >> 16) & 1u)) >> 16); // RNE
}
DEV unsigned pack2(float a, float b) {
  return (unsigned)f2bf(a) | ((unsigned)f2bf(b) << 16);
}

union FragU { uint4 q[2]; v16bf v; };

// A fragment, 16x32 bf16, row-major source with row stride ldm (elems).
// lanes 0-15 M=0-15 / 16-31 M=0-15; VGPR j: K = lh*16 + 2j (matches B fragments).
// => per lane: 32 contiguous bytes [lh*32, lh*32+32) of its row-window.
DEV v16bf frag_a_lds(const __bf16* base, int ldm) {
  int lane = threadIdx.x & 31;
  const uint4* p = (const uint4*)(base + (lane & 15) * ldm);
  int lh = lane >> 4;
  FragU f; f.q[0] = p[2 * lh]; f.q[1] = p[2 * lh + 1];
  return f.v;
}

// B fragment, 32x16 bf16, from B^T (rows = N, cols = K) with row stride ldk.
// B layout used here (contiguous K, not the ISA table order): lane n=lane%16, VGPR j holds K = lh*16 + 2j, 2j+1.
// => per lane: 32 contiguous bytes of row n at K-offset lh*16.
DEV v16bf frag_bT_lds(const __bf16* baseT, int ldk) {
  int lane = threadIdx.x & 31;
  const uint4* p = (const uint4*)(baseT + (lane & 15) * ldk + (lane >> 4) * 16);
  FragU f; f.q[0] = p[0]; f.q[1] = p[1];
  return f.v;
}

// B fragment from pre-swizzled global tile (32 lanes x 8 u32, coalesced).
DEV v16bf frag_b_swz(const unsigned* tp) {
  int lane = threadIdx.x & 31;
  const uint4* p = (const uint4*)(tp + lane * 8);
  FragU f; f.q[0] = p[0]; f.q[1] = p[1];
  return f.v;
}

DEV v8f wmma_bf(v16bf a, v16bf b, v8f c) {
  return __builtin_amdgcn_wmma_f32_16x16x32_bf16(false, a, false, b, (short)0, c,
                                                 false, false);
}

// ---------------- prep kernels ----------------

__global__ void k_cvt_x(const float* __restrict__ x, unsigned* __restrict__ xb) {
  size_t i = (size_t)blockIdx.x * blockDim.x + threadIdx.x; // one float4 each
  float4 f = ((const float4*)x)[i];
  uint2 o; o.x = pack2(f.x, f.y); o.y = pack2(f.z, f.w);
  ((uint2*)xb)[i] = o;
}

// Swizzle W [Kd, N] f32 -> B-fragment-ordered bf16 pairs.
// u32 g: j=g&7, lane=(g>>3)&31, tile=g>>8; kt=tile%(Kd/32), nt=tile/(Kd/32);
// value = pack(B[kt*32 + lh*16 + 2j, nt*16 + lane%16], next-k).
__global__ void k_swz_w(const float* __restrict__ w, unsigned* __restrict__ o,
                        int Kd, int N) {
  int g = blockIdx.x * blockDim.x + threadIdx.x;
  int j = g & 7, lane = (g >> 3) & 31, tile = g >> 8;
  int ktiles = Kd >> 5;
  int kt = tile % ktiles, nt = tile / ktiles;
  int n = nt * 16 + (lane & 15);
  int k = kt * 32 + (lane >> 4) * 16 + j * 2;
  o[g] = pack2(w[(size_t)k * N + n], w[(size_t)(k + 1) * N + n]);
}

__global__ void k_zero(float* p, int n4) {
  int i = blockIdx.x * blockDim.x + threadIdx.x;
  if (i < n4) ((float4*)p)[i] = make_float4(0.f, 0.f, 0.f, 0.f);
}

// ---------------- fused projection + LayerNorm ----------------
// Block: 64 rows x 128 cols (one head) of X @ W. doLN: per-row LN then store
// TRANSPOSED [d, L] bf16 (for K^T / V^T consumption); else Q stored [L, d].
__global__ void __launch_bounds__(256)
k_proj(const unsigned* __restrict__ xb, const unsigned* __restrict__ wswz,
       const float* __restrict__ bias, const float* __restrict__ gamma,
       const float* __restrict__ beta, unsigned short* __restrict__ dst,
       int doLN) {
  const int l0 = blockIdx.x * 64;
  const int head = blockIdx.y;
  const int b = blockIdx.z;
  const int t = threadIdx.x;

  __shared__ __align__(16) __bf16 xT[64 * 128];  // 16KB
  __shared__ __align__(16) float  oT[64 * 128];  // 32KB
  __shared__ float biasL[128], gL[128], bL[128], muL[64], rsL[64];

  { // contiguous 16KB copy of the X row-block
    const uint4* src = (const uint4*)xb + ((size_t)b * Ll + l0) * (Cin / 8);
    uint4* d4 = (uint4*)xT;
#pragma unroll
    for (int i = 0; i < 4; ++i) d4[t + i * 256] = src[t + i * 256];
  }
  if (t < 128) {
    biasL[t] = bias[head * 128 + t];
    gL[t] = doLN ? gamma[head * 128 + t] : 1.0f;
    bL[t] = doLN ? beta[head * 128 + t] : 0.0f;
  }
  __syncthreads();

  const int wv = t >> 5, rs = wv & 3, cg = wv >> 2; // 4 row-strips x 2 col-groups
  v8f acc[4] = {};
#pragma unroll
  for (int kk = 0; kk < 4; ++kk) {
    v16bf a = frag_a_lds(xT + rs * 16 * 128 + kk * 32, 128);
#pragma unroll
    for (int t4 = 0; t4 < 4; ++t4) {
      int ntile = head * 8 + cg * 4 + t4;
      acc[t4] = wmma_bf(a, frag_b_swz(wswz + (size_t)(ntile * 4 + kk) * 256), acc[t4]);
    }
  }
  { // spill C fragments to LDS f32 (ISA C layout: row = lh*8+r, col = lane%16)
    int lane = t & 31, lh = lane >> 4, ln = lane & 15;
#pragma unroll
    for (int t4 = 0; t4 < 4; ++t4)
#pragma unroll
      for (int r = 0; r < 8; ++r)
        oT[(rs * 16 + lh * 8 + r) * 128 + cg * 64 + t4 * 16 + ln] = acc[t4][r];
  }
  __syncthreads();

  if (doLN) {
    // LN stats: 4 lanes per row, partial sums + shfl_xor(1,2) within wave32
    int row = t >> 2, part = t & 3;
    float s = 0.f, ss = 0.f;
#pragma unroll
    for (int i = 0; i < 32; ++i) {
      int c = part * 32 + i;
      float y = oT[row * 128 + c] + biasL[c];
      s += y; ss += y * y;
    }
    s += __shfl_xor(s, 1, 32);  s += __shfl_xor(s, 2, 32);
    ss += __shfl_xor(ss, 1, 32); ss += __shfl_xor(ss, 2, 32);
    if (part == 0) {
      float mu = s * (1.f / 128.f);
      float var = ss * (1.f / 128.f) - mu * mu;
      muL[row] = mu; rsL[row] = rsqrtf(var + 1e-5f);
    }
    __syncthreads();
    // transposed bf16 store: dst[(bh*D + c)*L + l0 + li], coalesced over li
    unsigned short* base = dst + (size_t)(b * Hh + head) * Dd * Ll + l0;
#pragma unroll
    for (int i = 0; i < 32; ++i) {
      int g = t + i * 256;
      int c = g >> 6, li = g & 63;
      float y = oT[li * 128 + c] + biasL[c];
      y = (y - muL[li]) * rsL[li] * gL[c] + bL[c];
      base[(size_t)c * Ll + li] = f2bf(y);
    }
  } else {
    // Q: bias + bf16, row-major [L, d], packed u32 stores
    unsigned* qb = (unsigned*)dst;
    size_t rowbase = ((size_t)(b * Hh + head) * Ll + l0) * (Dd / 2);
#pragma unroll
    for (int i = 0; i < 16; ++i) {
      int g = t + i * 256;
      int row = g >> 6, cp = g & 63;
      float y0 = oT[row * 128 + 2 * cp] + biasL[2 * cp];
      float y1 = oT[row * 128 + 2 * cp + 1] + biasL[2 * cp + 1];
      qb[rowbase + (size_t)row * 64 + cp] = pack2(y0, y1);
    }
  }
}

// ---------------- M = K^T V, split-K over L, double-buffered LDS ----------------
__global__ void __launch_bounds__(256)
k_ktv(const unsigned short* __restrict__ Kt, const unsigned short* __restrict__ Vt,
      float* __restrict__ M) {
  const int chunk = blockIdx.x; // 16 chunks of 2048
  const int head = blockIdx.y, b = blockIdx.z;
  const int t = threadIdx.x;
  const size_t bh = (size_t)(b * Hh + head);
  const unsigned short* Kp = Kt + bh * Dd * Ll;
  const unsigned short* Vp = Vt + bh * Dd * Ll;
  const int lbase = chunk * (Ll / 16);
  constexpr int NKS = 64;

  __shared__ __align__(16) __bf16 kT[2][128 * 32]; // K^T slabs (rows=d, cols=l)
  __shared__ __align__(16) __bf16 vT[2][128 * 32]; // V^T slabs

  const int wv = t >> 5; // wave w -> M rows [w*16, w*16+16), all 8 col tiles
  v8f acc[8] = {};

  const int g0 = t, g1 = t + 256; // 512 uint4 per slab
  auto gld = [&](const unsigned short* P, int g, int loff) {
    int r = g >> 2, part = g & 3;
    return *((const uint4*)(P + (size_t)r * Ll + loff) + part);
  };

  // prologue: stage slab 0
  uint4 rk0 = gld(Kp, g0, lbase), rk1 = gld(Kp, g1, lbase);
  uint4 rv0 = gld(Vp, g0, lbase), rv1 = gld(Vp, g1, lbase);
  ((uint4*)kT[0])[g0] = rk0; ((uint4*)kT[0])[g1] = rk1;
  ((uint4*)vT[0])[g0] = rv0; ((uint4*)vT[0])[g1] = rv1;
  __syncthreads();

  for (int ks = 0; ks < NKS; ++ks) {
    const int cur = ks & 1, nxt = cur ^ 1;
    if (ks + 1 < NKS) {
      // issue next-slab global loads before the WMMA burst (latency overlap)
      const int lo = lbase + (ks + 1) * 32;
      rk0 = gld(Kp, g0, lo); rk1 = gld(Kp, g1, lo);
      rv0 = gld(Vp, g0, lo); rv1 = gld(Vp, g1, lo);
      if (ks + 2 < NKS) {
        // global_prefetch_b8: pull slab ks+2 into L2 early (streamed once)
        const int lo2 = lbase + (ks + 2) * 32;
        __builtin_prefetch(Kp + (size_t)(g0 >> 2) * Ll + lo2, 0, 3);
        __builtin_prefetch(Vp + (size_t)(g0 >> 2) * Ll + lo2, 0, 3);
      }
    }
    v16bf a = frag_a_lds(kT[cur] + wv * 16 * 32, 32);
#pragma unroll
    for (int nt = 0; nt < 8; ++nt)
      acc[nt] = wmma_bf(a, frag_bT_lds(vT[cur] + nt * 16 * 32, 32), acc[nt]);
    if (ks + 1 < NKS) {
      ((uint4*)kT[nxt])[g0] = rk0; ((uint4*)kT[nxt])[g1] = rk1;
      ((uint4*)vT[nxt])[g0] = rv0; ((uint4*)vT[nxt])[g1] = rv1;
      __syncthreads(); // single barrier per slab
    }
  }

  int lane = t & 31, lh = lane >> 4, ln = lane & 15;
  float* Mp = M + bh * Dd * Dd;
#pragma unroll
  for (int nt = 0; nt < 8; ++nt)
#pragma unroll
    for (int r = 0; r < 8; ++r)
      atomicAdd(&Mp[(wv * 16 + lh * 8 + r) * 128 + nt * 16 + ln], acc[nt][r]);
}

// ---------------- P_h = bf16(M_h) @ Wo_h, stored swizzled ----------------
__global__ void __launch_bounds__(256)
k_mwo(const float* __restrict__ M, const unsigned* __restrict__ woswz,
      unsigned* __restrict__ Pswz) {
  const int bh = blockIdx.x;
  const int head = bh % Hh;
  const int t = threadIdx.x;
  __shared__ __align__(16) float lsF[128 * 128]; // 64KB; aliased as bf16 first
  __bf16* Mb = (__bf16*)lsF;
  unsigned* Mb32 = (unsigned*)lsF;

  const float* Mp = M + (size_t)bh * Dd * Dd;
#pragma unroll
  for (int i = 0; i < 32; ++i) {
    int g = t + i * 256;                            // pair index
    Mb32[g] = pack2(Mp[2 * g], Mp[2 * g + 1]);
  }
  __syncthreads();

  const int wv = t >> 5;
  v8f acc[8] = {};
#pragma unroll
  for (int kk = 0; kk < 4; ++kk) {
    v16bf a = frag_a_lds(Mb + wv * 16 * 128 + kk * 32, 128);
#pragma unroll
    for (int nt = 0; nt < 8; ++nt)
      acc[nt] = wmma_bf(a, frag_b_swz(woswz + (size_t)(nt * 32 + head * 4 + kk) * 256),
                        acc[nt]);
  }
  __syncthreads(); // done with Mb; reuse LDS as f32 P staging
  int lane = t & 31, lh = lane >> 4, ln = lane & 15;
#pragma unroll
  for (int nt = 0; nt < 8; ++nt)
#pragma unroll
    for (int r = 0; r < 8; ++r)
      lsF[(wv * 16 + lh * 8 + r) * 128 + nt * 16 + ln] = acc[nt][r];
  __syncthreads();
  unsigned* Pp = Pswz + (size_t)bh * (Dd * Dd / 2);
#pragma unroll
  for (int i = 0; i < 32; ++i) {
    int g = t + i * 256;
    int j = g & 7, lane2 = (g >> 3) & 31, tile = g >> 8;
    int kt = tile & 3, nt = tile >> 2;
    int n = nt * 16 + (lane2 & 15);
    int k = kt * 32 + (lane2 >> 4) * 16 + j * 2;
    Pp[g] = pack2(lsF[k * 128 + n], lsF[(k + 1) * 128 + n]);
  }
}

// ---------------- out = (sum_h Q_h @ P_h + Wo_b) / n, double-buffered Q ----------
__global__ void __launch_bounds__(256)
k_out(const unsigned* __restrict__ Qb, const unsigned* __restrict__ Pswz,
      const float* __restrict__ WoB, float* __restrict__ out) {
  const int l0 = blockIdx.x * 64;
  const int b = blockIdx.y;
  const int t = threadIdx.x;
  __shared__ __align__(16) __bf16 qT[2][64 * 128]; // 32KB

  const int wv = t >> 5, rs = wv & 3, cg = wv >> 2;
  v8f acc[4] = {};

  auto qsrc = [&](int head) {
    return (const uint4*)Qb + ((size_t)(b * Hh + head) * Ll + l0) * (Dd / 8);
  };
  uint4 rq[4];
  { // prologue: stage head 0
    const uint4* src = qsrc(0);
    uint4* d4 = (uint4*)qT[0];
#pragma unroll
    for (int i = 0; i < 4; ++i) d4[t + i * 256] = src[t + i * 256];
  }
  __syncthreads();

  for (int head = 0; head < 8; ++head) {
    const int cur = head & 1, nxt = cur ^ 1;
    if (head + 1 < 8) {
      const uint4* src = qsrc(head + 1);
#pragma unroll
      for (int i = 0; i < 4; ++i) rq[i] = src[t + i * 256];
    }
    const unsigned* Pp = Pswz + (size_t)(b * Hh + head) * (Dd * Dd / 2);
#pragma unroll
    for (int kk = 0; kk < 4; ++kk) {
      v16bf a = frag_a_lds(qT[cur] + rs * 16 * 128 + kk * 32, 128);
#pragma unroll
      for (int t4 = 0; t4 < 4; ++t4) {
        int nt = cg * 4 + t4;
        acc[t4] = wmma_bf(a, frag_b_swz(Pp + (size_t)(nt * 4 + kk) * 256), acc[t4]);
      }
    }
    if (head + 1 < 8) {
      uint4* d4 = (uint4*)qT[nxt];
#pragma unroll
      for (int i = 0; i < 4; ++i) d4[t + i * 256] = rq[i];
      __syncthreads(); // single barrier per head
    }
  }
  const float invn = 1.0f / 1024.0f; // n = nx*ny
  int lane = t & 31, lh = lane >> 4, ln = lane & 15;
#pragma unroll
  for (int t4 = 0; t4 < 4; ++t4)
#pragma unroll
    for (int r = 0; r < 8; ++r) {
      int row = rs * 16 + lh * 8 + r;
      int col = cg * 64 + t4 * 16 + ln;
      out[((size_t)b * Ll + l0 + row) * Dd + col] = (acc[t4][r] + WoB[col]) * invn;
    }
}

// ---------------- host launch ----------------
extern "C" void kernel_launch(void* const* d_in, const int* in_sizes, int n_in,
                              void* d_out, int out_size, void* d_ws, size_t ws_size,
                              hipStream_t stream) {
  (void)in_sizes; (void)n_in; (void)out_size; (void)ws_size;
  const float* v    = (const float*)d_in[0];
  const float* WqW  = (const float*)d_in[1];
  const float* WqB  = (const float*)d_in[2];
  const float* WkW  = (const float*)d_in[3];
  const float* WkB  = (const float*)d_in[4];
  const float* WvW  = (const float*)d_in[5];
  const float* WvB  = (const float*)d_in[6];
  const float* WoW  = (const float*)d_in[7];
  const float* WoB  = (const float*)d_in[8];
  const float* lnkG = (const float*)d_in[9];
  const float* lnkB = (const float*)d_in[10];
  const float* lnvG = (const float*)d_in[11];
  const float* lnvB = (const float*)d_in[12];

  char* ws = (char*)d_ws;
  size_t off = 0;
  auto alloc = [&](size_t bytes) {
    void* p = ws + off; off += (bytes + 255) & ~(size_t)255; return p;
  };
  unsigned* xb        = (unsigned*)alloc((size_t)Bz * Ll * Cin * 2);
  unsigned* wqS       = (unsigned*)alloc((size_t)Cin * HDim * 2);
  unsigned* wkS       = (unsigned*)alloc((size_t)Cin * HDim * 2);
  unsigned* wvS       = (unsigned*)alloc((size_t)Cin * HDim * 2);
  unsigned* woS       = (unsigned*)alloc((size_t)HDim * Dd * 2);
  unsigned short* Qb  = (unsigned short*)alloc((size_t)Bz * Hh * Ll * Dd * 2);
  unsigned short* Kt  = (unsigned short*)alloc((size_t)Bz * Hh * Dd * Ll * 2);
  unsigned short* Vt  = (unsigned short*)alloc((size_t)Bz * Hh * Dd * Ll * 2);
  float* M            = (float*)alloc((size_t)Bz * Hh * Dd * Dd * 4);
  unsigned* Pswz      = (unsigned*)alloc((size_t)Bz * Hh * Dd * Dd * 2);

  k_cvt_x<<<(Bz * Ll * Cin) / (256 * 4), 256, 0, stream>>>(v, xb);
  k_swz_w<<<(Cin * HDim / 2) / 256, 256, 0, stream>>>(WqW, wqS, Cin, HDim);
  k_swz_w<<<(Cin * HDim / 2) / 256, 256, 0, stream>>>(WkW, wkS, Cin, HDim);
  k_swz_w<<<(Cin * HDim / 2) / 256, 256, 0, stream>>>(WvW, wvS, Cin, HDim);
  k_swz_w<<<(HDim * Dd / 2) / 256, 256, 0, stream>>>(WoW, woS, HDim, Dd);
  k_zero<<<(Bz * Hh * Dd * Dd / 4) / 256, 256, 0, stream>>>(M, Bz * Hh * Dd * Dd / 4);

  dim3 gProj(Ll / 64, Hh, Bz);
  k_proj<<<gProj, 256, 0, stream>>>(xb, wqS, WqB, nullptr, nullptr, Qb, 0);
  k_proj<<<gProj, 256, 0, stream>>>(xb, wkS, WkB, lnkG, lnkB, Kt, 1);
  k_proj<<<gProj, 256, 0, stream>>>(xb, wvS, WvB, lnvG, lnvB, Vt, 1);

  k_ktv<<<dim3(16, Hh, Bz), 256, 0, stream>>>(Kt, Vt, M);
  k_mwo<<<Bz * Hh, 256, 0, stream>>>(M, woS, Pswz);
  k_out<<<dim3(Ll / 64, Bz), 256, 0, stream>>>((const unsigned*)Qb, Pswz, WoB,
                                               (float*)d_out);
}
